// GPT2ModelBlock_60430189854901
// MI455X (gfx1250) — matrix operand order repaired and checked by reading, not compiled, not run on hardware
//
#include <hip/hip_runtime.h>
#include <cstdint>
#include <cstddef>

// ---------------- problem constants ----------------
#define BB 2
#define SS 2048
#define EE 1024
#define HH 16
#define DD 64
#define FF 4096
#define MM (BB*SS)          // 4096 token rows

typedef __attribute__((ext_vector_type(16))) __bf16 v16bf;
typedef __attribute__((ext_vector_type(8)))  __bf16 v8bf;
typedef __attribute__((ext_vector_type(8)))  float  v8f;

static __device__ __forceinline__ v8f wmma_bf16(v16bf a, v16bf b, v8f c) {
  // D = A(16x32 bf16) * B(32x16 bf16) + C(16x16 f32)
  return __builtin_amdgcn_wmma_f32_16x16x32_bf16(false, a, false, b, (short)0, c,
                                                 false, false);
}

// CDNA5 async global->LDS copy (ASYNCcnt-tracked DMA path, no VGPR data).
// lds_off: byte offset within the workgroup's LDS allocation (VDST VGPR).
static __device__ __forceinline__ void async_ld16(unsigned lds_off,
                                                  unsigned long long gaddr) {
  asm volatile("global_load_async_to_lds_b128 %0, %1, off"
               :: "v"(lds_off), "v"(gaddr)
               : "memory");
}
static __device__ __forceinline__ void wait_async0() {
  asm volatile("s_wait_asynccnt 0" ::: "memory");
}

// ---------------- f32 -> bf16 convert ----------------
__global__ __launch_bounds__(256)
void cvt_bf16_kernel(const float* __restrict__ in, __bf16* __restrict__ out, int n) {
  int i = blockIdx.x * 256 + threadIdx.x;
  if (i < n) out[i] = (__bf16)in[i];
}

// ---------------- LayerNorm (row = one block) ----------------
__global__ __launch_bounds__(256)
void layernorm_kernel(const float* __restrict__ X, const float* __restrict__ w,
                      const float* __restrict__ b, float* __restrict__ outF,
                      __bf16* __restrict__ outB) {
  const int row = blockIdx.x;
  const int tid = threadIdx.x;
  const float* xr = X + (size_t)row * EE;

  float s = 0.f, q = 0.f;
  #pragma unroll
  for (int i = tid; i < EE; i += 256) { float v = xr[i]; s += v; q += v * v; }

  __shared__ float sb[256], qb[256];
  sb[tid] = s; qb[tid] = q;
  __syncthreads();
  for (int o2 = 128; o2 > 0; o2 >>= 1) {
    if (tid < o2) { sb[tid] += sb[tid + o2]; qb[tid] += qb[tid + o2]; }
    __syncthreads();
  }
  const float mean = sb[0] * (1.0f / EE);
  const float var  = qb[0] * (1.0f / EE) - mean * mean;
  const float rstd = rsqrtf(var + 1e-5f);

  #pragma unroll
  for (int i = tid; i < EE; i += 256) {
    float y = (xr[i] - mean) * rstd * w[i] + b[i];
    if (outF) outF[(size_t)row * EE + i] = y;
    outB[(size_t)row * EE + i] = (__bf16)y;
  }
}

// ---------------- block-tiled WMMA GEMM with async-LDS double buffering ----------
// C[M,N] = A[M,K] * W[N,K]^T (+bias, +gelu, +resid). Block tile 128x128, BK=32.
// 8 waves, each owns a 64(M) x 32(N) sub-tile = 4x2 WMMA frags.
// Global->LDS staging via GLOBAL_LOAD_ASYNC_TO_LDS_B128 (ASYNCcnt), double buffered.
__global__ __launch_bounds__(256)
void gemm_bf16_wmma(const __bf16* __restrict__ A, const __bf16* __restrict__ W,
                    const float* __restrict__ bias, const float* __restrict__ resid,
                    float* __restrict__ outF, __bf16* __restrict__ outB,
                    int M, int N, int K, int act) {
  const int tid  = threadIdx.x;
  const int lane = tid & 31;
  const int wave = tid >> 5;
  const int half = lane >> 4;
  const int lm   = lane & 15;
  const int wmB  = (wave >> 2) * 64;   // wave M base within block tile
  const int wnB  = (wave & 3) * 32;    // wave N base within block tile
  const int m0b  = blockIdx.y * 128;
  const int n0b  = blockIdx.x * 128;

  // LDS: double-buffered A (128x32) and W (128x32) bf16 tiles, 8KB each.
  __shared__ alignas(16) __bf16 Ash[2][128 * 32];
  __shared__ alignas(16) __bf16 Bsh[2][128 * 32];
  const unsigned aOff[2] = { (unsigned)(size_t)&Ash[0][0], (unsigned)(size_t)&Ash[1][0] };
  const unsigned bOff[2] = { (unsigned)(size_t)&Bsh[0][0], (unsigned)(size_t)&Bsh[1][0] };

  // stage k-tile kt into buffer buf: 512 16B chunks each for A and W, 2+2 per thread
  auto stage = [&](int buf, int kt) {
    const int k0 = kt * 32;
    #pragma unroll
    for (int c2 = 0; c2 < 2; ++c2) {
      const int chunk = tid * 2 + c2;        // 0..511
      const int r = chunk >> 2;              // row 0..127
      const int c = chunk & 3;               // 16B chunk in 64B row
      const unsigned l = (unsigned)(r * 64 + c * 16);
      async_ld16(aOff[buf] + l,
                 (unsigned long long)(A + (size_t)(m0b + r) * K + k0 + c * 8));
      async_ld16(bOff[buf] + l,
                 (unsigned long long)(W + (size_t)(n0b + r) * K + k0 + c * 8));
    }
  };

  v8f acc[4][2];
  #pragma unroll
  for (int i = 0; i < 4; i++)
    #pragma unroll
    for (int j = 0; j < 2; j++)
      #pragma unroll
      for (int e = 0; e < 8; e++) acc[i][j][e] = 0.0f;

  const int nk = K / 32;
  stage(0, 0);
  wait_async0();
  __syncthreads();

  for (int kt = 0; kt < nk; ++kt) {
    const int buf = kt & 1;
    if (kt + 1 < nk) stage(buf ^ 1, kt + 1);

    const v8bf* Ald = (const v8bf*)&Ash[buf][0];   // 4 chunks per 32-elem row
    const v8bf* Bld = (const v8bf*)&Bsh[buf][0];

    v16bf af[4], bfr[2];
    #pragma unroll
    for (int i = 0; i < 4; i++) {
      const int m = wmB + i * 16 + lm;             // A lane row = M%16
      v8bf lo = Ald[m * 4 + half];                 // K chunk 8*half..+7
      v8bf hi = Ald[m * 4 + 2 + half];             // K chunk 16+8*half..+7
      #pragma unroll
      for (int e = 0; e < 8; e++) { af[i][e] = lo[e]; af[i][8 + e] = hi[e]; }
    }
    #pragma unroll
    for (int j = 0; j < 2; j++) {
      const int n = wnB + j * 16 + lm;             // B lane col = N%16
      v8bf lo = Bld[n * 4 + half];                 // K chunk 8*half..+7
      v8bf hi = Bld[n * 4 + 2 + half];             // K chunk 16+8*half..+7
      #pragma unroll
      for (int e = 0; e < 8; e++) { bfr[j][e] = lo[e]; bfr[j][8 + e] = hi[e]; }
    }
    #pragma unroll
    for (int i = 0; i < 4; i++)
      #pragma unroll
      for (int j = 0; j < 2; j++)
        acc[i][j] = wmma_bf16(af[i], bfr[j], acc[i][j]);

    wait_async0();       // our wave's async stores to LDS (next tile) complete
    __syncthreads();     // all waves' reads of current tile + stages complete
  }

  // epilogue
  #pragma unroll
  for (int j = 0; j < 2; j++) {
    const int n = n0b + wnB + j * 16 + lm;
    const float bv = bias ? bias[n] : 0.0f;
    #pragma unroll
    for (int i = 0; i < 4; i++) {
      #pragma unroll
      for (int e = 0; e < 8; e++) {
        const int m = m0b + wmB + i * 16 + e + 8 * half;   // C row = vgpr + 8*half
        float v = acc[i][j][e] + bv;
        if (act == 1) v = 0.5f * v * (1.0f + erff(v * 0.70710678118654752f));
        if (resid) v += resid[(size_t)m * N + n];
        if (outF) outF[(size_t)m * N + n] = v;
        if (outB) outB[(size_t)m * N + n] = (__bf16)v;
      }
    }
  }
}

// ---------------- causal flash attention, WMMA, one wave per (b,h,q-tile of 16) ------
__global__ __launch_bounds__(32)
void attn_flash_wmma(const __bf16* __restrict__ Q, const __bf16* __restrict__ Kb,
                     const __bf16* __restrict__ Vb, __bf16* __restrict__ O) {
  const int lane = threadIdx.x & 31;
  const int half = lane >> 4;
  const int lm   = lane & 15;
  const int qt = blockIdx.x;     // 0..S/16-1
  const int h  = blockIdx.y;
  const int b  = blockIdx.z;
  const int q0 = qt * 16;

  __shared__ __bf16 Pl[16 * 32];   // P tile staging (C-layout -> A-layout)

  // Q fragments (K-dim = D = 64 -> two 16x32 A-frags)
  const __bf16* qr = Q + ((size_t)(b * SS + q0 + lm) * HH + h) * (size_t)DD;
  v16bf qa[2];
  #pragma unroll
  for (int f = 0; f < 2; f++) {
    v8bf lo = *(const v8bf*)(qr + 32 * f + 8 * half);
    v8bf hi = *(const v8bf*)(qr + 32 * f + 8 * half + 16);
    #pragma unroll
    for (int e = 0; e < 8; e++) { qa[f][e] = lo[e]; qa[f][8 + e] = hi[e]; }
  }

  float mrow[8], lrow[8];
  v8f o_acc[4];
  #pragma unroll
  for (int i = 0; i < 8; i++) { mrow[i] = -1.0e30f; lrow[i] = 0.0f; }
  #pragma unroll
  for (int dt = 0; dt < 4; dt++)
    #pragma unroll
    for (int i = 0; i < 8; i++) o_acc[dt][i] = 0.0f;

  const int nt = (q0 + 15) / 32 + 1;     // causal: kv tiles of 32
  for (int t = 0; t < nt; t++) {
    const int kv0 = 32 * t;

    // ---- scores: two 16-wide column sets, each = 2 WMMAs over d ----
    const __bf16* krA = Kb + ((size_t)(b * SS + kv0 + lm)      * HH + h) * (size_t)DD;
    const __bf16* krB = Kb + ((size_t)(b * SS + kv0 + 16 + lm) * HH + h) * (size_t)DD;
    v8f sA, sB;
    #pragma unroll
    for (int i = 0; i < 8; i++) { sA[i] = 0.0f; sB[i] = 0.0f; }
    #pragma unroll
    for (int f = 0; f < 2; f++) {
      v8bf lo = *(const v8bf*)(krA + 32 * f + 8 * half);
      v8bf hi = *(const v8bf*)(krA + 32 * f + 8 * half + 16);
      v16bf kf;
      #pragma unroll
      for (int e = 0; e < 8; e++) { kf[e] = lo[e]; kf[8 + e] = hi[e]; }
      sA = wmma_bf16(qa[f], kf, sA);
    }
    #pragma unroll
    for (int f = 0; f < 2; f++) {
      v8bf lo = *(const v8bf*)(krB + 32 * f + 8 * half);
      v8bf hi = *(const v8bf*)(krB + 32 * f + 8 * half + 16);
      v16bf kf;
      #pragma unroll
      for (int e = 0; e < 8; e++) { kf[e] = lo[e]; kf[8 + e] = hi[e]; }
      sB = wmma_bf16(qa[f], kf, sB);
    }

    // ---- causal mask + online softmax (rows live in 16-lane half-groups) ----
    float pA[8], pB[8];
    #pragma unroll
    for (int i = 0; i < 8; i++) {
      const int qrow = q0 + i + 8 * half;
      sA[i] = (kv0 + lm      <= qrow) ? sA[i] * 0.125f : -1.0e30f;  // 1/sqrt(64)
      sB[i] = (kv0 + 16 + lm <= qrow) ? sB[i] * 0.125f : -1.0e30f;
      float mx = fmaxf(sA[i], sB[i]);
      for (int s2 = 8; s2 >= 1; s2 >>= 1) mx = fmaxf(mx, __shfl_xor(mx, s2, 32));
      const float mnew  = fmaxf(mrow[i], mx);
      const float alpha = __expf(mrow[i] - mnew);
      pA[i] = __expf(sA[i] - mnew);
      pB[i] = __expf(sB[i] - mnew);
      float rs = pA[i] + pB[i];
      for (int s2 = 8; s2 >= 1; s2 >>= 1) rs += __shfl_xor(rs, s2, 32);
      lrow[i] = lrow[i] * alpha + rs;
      mrow[i] = mnew;
      #pragma unroll
      for (int dt = 0; dt < 4; dt++) o_acc[dt][i] *= alpha;
    }

    // ---- stage P (C-layout) to LDS, reload as A-frag (16x32) ----
    #pragma unroll
    for (int i = 0; i < 8; i++) {
      Pl[(i + 8 * half) * 32 + lm]      = (__bf16)pA[i];
      Pl[(i + 8 * half) * 32 + 16 + lm] = (__bf16)pB[i];
    }
    __syncthreads();                 // single-wave WG: orders DS ops
    v16bf pfrag;
    #pragma unroll
    for (int e = 0; e < 8; e++) {
      pfrag[e]     = Pl[lm * 32 + 8 * half + e];
      pfrag[8 + e] = Pl[lm * 32 + 16 + 8 * half + e];
    }
    __syncthreads();

    // ---- O += P * V  (4 d-tiles of 16) ----
    #pragma unroll
    for (int dt = 0; dt < 4; dt++) {
      v16bf vf;
      #pragma unroll
      for (int e = 0; e < 16; e++) {
        const int kv = kv0 + (e < 8 ? 8 * half + e : 16 + 8 * half + (e - 8));
        vf[e] = Vb[((size_t)(b * SS + kv) * HH + h) * (size_t)DD + dt * 16 + lm];
      }
      o_acc[dt] = wmma_bf16(pfrag, vf, o_acc[dt]);
    }
  }

  // ---- epilogue: normalize and store bf16 O in [b,s,h,d] ----
  #pragma unroll
  for (int i = 0; i < 8; i++) {
    const float inv = 1.0f / lrow[i];
    const size_t ro = ((size_t)(b * SS + q0 + i + 8 * half) * HH + h) * (size_t)DD;
    #pragma unroll
    for (int dt = 0; dt < 4; dt++)
      O[ro + dt * 16 + lm] = (__bf16)(o_acc[dt][i] * inv);
  }
}

// ---------------- host orchestration ----------------
extern "C" void kernel_launch(void* const* d_in, const int* in_sizes, int n_in,
                              void* d_out, int out_size, void* d_ws, size_t ws_size,
                              hipStream_t stream) {
  (void)in_sizes; (void)n_in; (void)out_size; (void)ws_size;
  const float* x    = (const float*)d_in[0];
  const float* ln1w = (const float*)d_in[1];
  const float* ln1b = (const float*)d_in[2];
  const float* ln2w = (const float*)d_in[3];
  const float* ln2b = (const float*)d_in[4];
  const float* wq_w = (const float*)d_in[5];
  const float* wq_b = (const float*)d_in[6];
  const float* wk_w = (const float*)d_in[7];
  const float* wk_b = (const float*)d_in[8];
  const float* wv_w = (const float*)d_in[9];
  const float* wv_b = (const float*)d_in[10];
  const float* wo_w = (const float*)d_in[11];
  const float* wo_b = (const float*)d_in[12];
  const float* up_w = (const float*)d_in[13];
  const float* up_b = (const float*)d_in[14];
  const float* dn_w = (const float*)d_in[15];
  const float* dn_b = (const float*)d_in[16];
  float* out = (float*)d_out;

  // workspace carve-up (~136 MB)
  char* ws = (char*)d_ws;
  size_t off = 0;
  auto carve = [&](size_t bytes) -> char* {
    char* p = ws + off;
    off += (bytes + 255) & ~(size_t)255;
    return p;
  };
  __bf16* x1b    = (__bf16*)carve((size_t)MM * EE * 2);
  __bf16* qb     = (__bf16*)carve((size_t)MM * EE * 2);
  __bf16* kb     = (__bf16*)carve((size_t)MM * EE * 2);
  __bf16* vb     = (__bf16*)carve((size_t)MM * EE * 2);
  __bf16* ob     = (__bf16*)carve((size_t)MM * EE * 2);
  float*  xat    = (float*) carve((size_t)MM * EE * 4);
  float*  xln2f  = (float*) carve((size_t)MM * EE * 4);
  __bf16* xln2b  = (__bf16*)carve((size_t)MM * EE * 2);
  __bf16* hb     = (__bf16*)carve((size_t)MM * FF * 2);
  __bf16* wqb16  = (__bf16*)carve((size_t)EE * EE * 2);
  __bf16* wkb16  = (__bf16*)carve((size_t)EE * EE * 2);
  __bf16* wvb16  = (__bf16*)carve((size_t)EE * EE * 2);
  __bf16* wob16  = (__bf16*)carve((size_t)EE * EE * 2);
  __bf16* upb16  = (__bf16*)carve((size_t)FF * EE * 2);
  __bf16* dnb16  = (__bf16*)carve((size_t)EE * FF * 2);

  auto cvt = [&](const float* src, __bf16* dst, int n) {
    cvt_bf16_kernel<<<(n + 255) / 256, 256, 0, stream>>>(src, dst, n);
  };
  cvt(wq_w, wqb16, EE * EE);
  cvt(wk_w, wkb16, EE * EE);
  cvt(wv_w, wvb16, EE * EE);
  cvt(wo_w, wob16, EE * EE);
  cvt(up_w, upb16, FF * EE);
  cvt(dn_w, dnb16, EE * FF);

  // LN1 -> bf16 activations
  layernorm_kernel<<<MM, 256, 0, stream>>>(x, ln1w, ln1b, nullptr, x1b);

  // Q/K/V projections (M=4096, N=1024, K=1024), block tile 128x128
  dim3 gQKV(EE / 128, MM / 128);
  gemm_bf16_wmma<<<gQKV, 256, 0, stream>>>(x1b, wqb16, wq_b, nullptr, nullptr, qb,
                                           MM, EE, EE, 0);
  gemm_bf16_wmma<<<gQKV, 256, 0, stream>>>(x1b, wkb16, wk_b, nullptr, nullptr, kb,
                                           MM, EE, EE, 0);
  gemm_bf16_wmma<<<gQKV, 256, 0, stream>>>(x1b, wvb16, wv_b, nullptr, nullptr, vb,
                                           MM, EE, EE, 0);

  // causal flash attention
  dim3 gAtt(SS / 16, HH, BB);
  attn_flash_wmma<<<gAtt, 32, 0, stream>>>(qb, kb, vb, ob);

  // WO projection + residual(x) -> f32
  gemm_bf16_wmma<<<gQKV, 256, 0, stream>>>(ob, wob16, wo_b, x, xat, nullptr,
                                           MM, EE, EE, 0);

  // LN2 -> f32 (final residual) + bf16 (MLP input)
  layernorm_kernel<<<MM, 256, 0, stream>>>(xat, ln2w, ln2b, xln2f, xln2b);

  // Up projection + exact GELU (M=4096, N=4096, K=1024)
  dim3 gUp(FF / 128, MM / 128);
  gemm_bf16_wmma<<<gUp, 256, 0, stream>>>(xln2b, upb16, up_b, nullptr, nullptr, hb,
                                          MM, FF, EE, 1);

  // Down projection + bias + residual(LN2 out) -> d_out (M=4096, N=1024, K=4096)
  dim3 gDn(EE / 128, MM / 128);
  gemm_bf16_wmma<<<gDn, 256, 0, stream>>>(hb, dnb16, dn_b, xln2f, out, nullptr,
                                          MM, EE, FF, 0);
}
